// PNA_28484223108047
// MI455X (gfx1250) — matrix operand order repaired and checked by reading, not compiled, not run on hardware
//
#include <hip/hip_runtime.h>
#include <hip/hip_bf16.h>

// ---------------------------------------------------------------------------
// PNA GNN for MI455X (gfx1250). All three GEMMs (edge MLP, U, mix) run on
// V_WMMA_F32_16X16X32_F16 (f16 in, f32 accumulate). Segment reductions are
// float atomics into L2-resident node accumulators (everything hot < 192MB L2).
// Shared B-operand weights are staged into LDS per block via the gfx1250
// async global->LDS path (ASYNCcnt) so the WMMA loop reads B from LDS.
// ---------------------------------------------------------------------------

typedef __attribute__((ext_vector_type(16))) _Float16 v16h;
typedef __attribute__((ext_vector_type(8)))  _Float16 v8h;
typedef __attribute__((ext_vector_type(8)))  float    v8f;
typedef __attribute__((ext_vector_type(4)))  int      v4i;

typedef __attribute__((address_space(1))) v4i* gv4i_p;   // global (prints as __device__)
typedef __attribute__((address_space(3))) v4i* lv4i_p;   // LDS    (prints as __shared__)

#define SLOPE   0.01f
#define DELTA_C 2.833213344056216f   // log(17)

#if defined(__gfx1250__) && __has_builtin(__builtin_amdgcn_global_load_async_to_lds_b128)
#define HAVE_ASYNC_LDS 1
#else
#define HAVE_ASYNC_LDS 0
#endif

static __device__ __forceinline__ float leaky(float v) { return v > 0.f ? v : v * SLOPE; }

// float atomic max/min via signed/unsigned int atomics.
// Buffers must be initialized to -INF (max) / +INF (min).
static __device__ __forceinline__ void atomMaxF(float* a, float v) {
    if (v >= 0.f) atomicMax((int*)a, __float_as_int(v));
    else          atomicMin((unsigned int*)a, __float_as_uint(v));
}
static __device__ __forceinline__ void atomMinF(float* a, float v) {
    if (v >= 0.f) atomicMin((int*)a, __float_as_int(v));
    else          atomicMax((unsigned int*)a, __float_as_uint(v));
}

static __device__ __forceinline__ v16h wmma_cat(v8h lo, v8h hi) {
    return __builtin_shufflevector(lo, hi, 0,1,2,3,4,5,6,7,8,9,10,11,12,13,14,15);
}

// Stage `halves` f16 elements (multiple of 8*threads) from global to LDS.
// All threads of the block must participate; caller must barrier afterwards.
static __device__ __forceinline__ void
stage_to_lds(_Float16* lds_dst, const _Float16* gsrc, int halves, int tid, int nthreads) {
#if HAVE_ASYNC_LDS
    for (int idx = tid * 8; idx < halves; idx += nthreads * 8) {
        __builtin_amdgcn_global_load_async_to_lds_b128(
            (gv4i_p)(gsrc + idx), (lv4i_p)(lds_dst + idx), 0, 0);
    }
#if __has_builtin(__builtin_amdgcn_s_wait_asynccnt)
    __builtin_amdgcn_s_wait_asynccnt(0);
#else
    asm volatile("s_wait_asynccnt 0" ::: "memory");
#endif
#else
    for (int idx = tid * 8; idx < halves; idx += nthreads * 8)
        *(v8h*)(lds_dst + idx) = *(const v8h*)(gsrc + idx);
#endif
}

// ------------------------------- small kernels ------------------------------

__global__ void k_round(const float* __restrict__ in, float* __restrict__ out, int n) {
    int i = blockIdx.x * blockDim.x + threadIdx.x;
    if (i < n) out[i] = rintf(in[i] * 100.f) / 100.f;   // jnp.round (RNE)
}

__global__ void k_fill(float* __restrict__ p, float v, int n) {
    int i = blockIdx.x * blockDim.x + threadIdx.x;
    if (i < n) p[i] = v;
}

__global__ void k_deg(const int* __restrict__ dst, float* __restrict__ deg, int E) {
    int i = blockIdx.x * blockDim.x + threadIdx.x;
    if (i < E) atomicAdd(&deg[dst[i]], 1.0f);
}

__global__ void k_scal(const float* __restrict__ deg, float* __restrict__ invd,
                       float* __restrict__ sa, float* __restrict__ sb, int N) {
    int i = blockIdx.x * blockDim.x + threadIdx.x;
    if (i >= N) return;
    float dg = deg[i];
    float dc = fmaxf(dg, 1.f);
    invd[i]  = 1.f / dc;
    float ld = logf(dc + 1.f);
    sa[i] = ld / DELTA_C;
    sb[i] = DELTA_C / ld;
}

__global__ void k_layer_init(const float* __restrict__ h, _Float16* __restrict__ h16,
                             float* __restrict__ sum, float* __restrict__ ssq,
                             float* __restrict__ mx, float* __restrict__ mn, int n) {
    int i = blockIdx.x * blockDim.x + threadIdx.x;
    if (i >= n) return;
    h16[i] = (_Float16)h[i];
    sum[i] = 0.f;
    ssq[i] = 0.f;
    mx[i]  = -__builtin_huge_valf();
    mn[i]  =  __builtin_huge_valf();
}

// Pack weights (all 3 layers) into WMMA B-operand order:
// B tile is 32(K) x 16(N); lane n holds column N=(n&15), h=(n>>4): halves 0..7 = K 8h..8h+7,
// halves 8..15 = K 16+8h..16+8h+7. Per (chunk c, ntile t): 32 lanes * 16 halves.
__global__ void k_pack(const float* __restrict__ Mw, const float* __restrict__ Uw,
                       const float* __restrict__ Xw, _Float16* __restrict__ pbM,
                       _Float16* __restrict__ pbU, _Float16* __restrict__ pbX) {
    int gid = blockIdx.x * blockDim.x + threadIdx.x;
    if (gid >= 3 * 32 * 4 * 32) return;          // 3 layers * 32 chunk-slots * 4 tiles * 32 lanes
    int lane  = gid & 31;
    int t     = (gid >> 5) & 3;
    int rest  = gid >> 7;
    int slot  = rest & 31;                        // 0..3 -> M, 4..29 -> U, 30..31 -> X
    int layer = rest >> 5;
    int n     = t * 16 + (lane & 15);
    int kb    = (lane & 16) ? 8 : 0;
    if (slot < 4) {
        int c = slot;
        const float* W = Mw + layer * 64 * 128;
        _Float16* d = pbM + (((layer * 4 + c) * 4 + t) * 32 + lane) * 16;
        #pragma unroll
        for (int k = 0; k < 16; ++k) d[k] = (_Float16)W[n * 128 + c * 32 + kb + k + (k & 8)];
    } else if (slot < 30) {
        int c = slot - 4;
        const float* W = Uw + layer * 64 * 832;
        _Float16* d = pbU + (((layer * 26 + c) * 4 + t) * 32 + lane) * 16;
        #pragma unroll
        for (int k = 0; k < 16; ++k) d[k] = (_Float16)W[n * 832 + c * 32 + kb + k + (k & 8)];
    } else {
        int c = slot - 30;
        const float* W = Xw + layer * 64 * 64;
        _Float16* d = pbX + (((layer * 2 + c) * 4 + t) * 32 + lane) * 16;
        #pragma unroll
        for (int k = 0; k < 16; ++k) d[k] = (_Float16)W[n * 64 + c * 32 + kb + k + (k & 8)];
    }
}

// ------------------------------ edge message kernel -------------------------
// One wave = 16 edges. msg[16x64] = leaky(F[16x128] @ Mw^T + b) via 16 WMMAs,
// then scatter sum/sumsq/max/min atomics to dst nodes. The 16KB of packed
// Mw B-tiles is staged into LDS once per block (async global->LDS).
__global__ __launch_bounds__(256) void
k_edge(const int* __restrict__ src, const int* __restrict__ dst,
       const _Float16* __restrict__ h16, const _Float16* __restrict__ pbM,
       const float* __restrict__ mb,
       float* __restrict__ sums, float* __restrict__ ssq,
       float* __restrict__ mxb, float* __restrict__ mnb, int E) {
    __shared__ __align__(16) _Float16 bS[16 * 32 * 16];   // 16KB: 16 B tiles
    stage_to_lds(bS, pbM, 16 * 32 * 16, threadIdx.x, 256);
    __syncthreads();

    int tid  = blockIdx.x * blockDim.x + threadIdx.x;
    int wid  = tid >> 5;
    int lane = threadIdx.x & 31;
    int base = wid * 16;
    if (base >= E) return;                       // wave-uniform, after barrier
    int  col = lane & 15;
    bool hig = (lane & 16) != 0;
    int  kk  = hig ? 8 : 0;                      // A-layout K sub-offset per lane half
    int  e   = base + col; if (e >= E) e = E - 1;
    const _Float16* rs = h16 + (long)src[e] * 64;
    const _Float16* rd = h16 + (long)dst[e] * 64;

    v8f acc[4] = {};
    #pragma unroll
    for (int c = 0; c < 4; ++c) {                // K chunks of 32 over 2D=128
        const _Float16* row = (c < 2) ? rs : rd;
        int kb = ((c & 1) << 5) + kk;
        v8h lo = *(const v8h*)(row + kb);
        v8h hi = *(const v8h*)(row + kb + 16);
        v16h a = wmma_cat(lo, hi);
        #pragma unroll
        for (int t = 0; t < 4; ++t) {
            v16h b = *(const v16h*)(bS + (((c << 2) | t) * 32 + lane) * 16);
            acc[t] = __builtin_amdgcn_wmma_f32_16x16x32_f16(
                false, a, false, b, (short)0, acc[t], false, false);
        }
    }

    float b4[4];
    #pragma unroll
    for (int t = 0; t < 4; ++t) b4[t] = mb[t * 16 + col];

    #pragma unroll
    for (int r = 0; r < 8; ++r) {                // D layout: lane holds (M = r+8*hig, N = col+16t)
        int m = base + r + (hig ? 8 : 0);
        if (m >= E) continue;
        long nb = (long)dst[m] * 64;
        #pragma unroll
        for (int t = 0; t < 4; ++t) {
            int   cn = t * 16 + col;
            float v  = leaky(acc[t][r] + b4[t]);
            atomicAdd(sums + nb + cn, v);
            atomicAdd(ssq  + nb + cn, v * v);
            atomMaxF(mxb + nb + cn, v);
            atomMinF(mnb + nb + cn, v);
        }
    }
}

// ------------------------------- node update kernel -------------------------
// One wave = 16 nodes. Builds 832-wide PNA feature rows on the fly, U-GEMM via
// 104 WMMAs, stages out1 through LDS to re-layout, mix-GEMM via 8 WMMAs
// (mix B-tiles staged in LDS), leaky + residual + relu.
static __device__ __forceinline__ void
build8(float* o, int k0, const float* hrow, const float* srow, const float* qrow,
       const float* xrow, const float* nrow, float invd, float saV, float sbV, float hasf) {
    if (k0 < 64) {
        #pragma unroll
        for (int i = 0; i < 8; ++i) o[i] = hrow[k0 + i];
        return;
    }
    int g  = (k0 - 64) >> 6;
    int j0 = (k0 - 64) & 63;
    int at = g & 3;                               // 0=mean 1=max 2=min 3=std
    int st = g >> 2;                              // 0=id 1=amp 2=atten
    float scl = (st == 0) ? 1.f : (st == 1 ? saV : sbV);
    #pragma unroll
    for (int i = 0; i < 8; ++i) {
        int j = j0 + i;
        float v;
        if (at == 0) v = srow[j] * invd;
        else if (at == 1) v = (hasf != 0.f) ? xrow[j] : 0.f;
        else if (at == 2) v = (hasf != 0.f) ? nrow[j] : 0.f;
        else {
            float mean = srow[j] * invd;
            float var  = fmaxf(qrow[j] * invd - mean * mean, 0.f);
            v = (hasf != 0.f) ? sqrtf(var + 1e-30f) : 0.f;
        }
        o[i] = v * scl;
    }
}

__global__ __launch_bounds__(256) void
k_node(const float* __restrict__ hcur, float* __restrict__ hnext,
       const float* __restrict__ sums, const float* __restrict__ ssq,
       const float* __restrict__ mxb, const float* __restrict__ mnb,
       const float* __restrict__ deg, const float* __restrict__ invd,
       const float* __restrict__ sa, const float* __restrict__ sb,
       const _Float16* __restrict__ pbU, const _Float16* __restrict__ pbX,
       const float* __restrict__ ub, const float* __restrict__ xb, int N) {
    __shared__ __align__(16) _Float16 lds[8 * 16 * 64];   // 16KB: per-wave 16x64 f16 stage
    __shared__ __align__(16) _Float16 bXs[8 * 32 * 16];   // 8KB: 8 mix B tiles
    stage_to_lds(bXs, pbX, 8 * 32 * 16, threadIdx.x, 256);   // barrier below covers this

    int tid  = blockIdx.x * blockDim.x + threadIdx.x;
    int wid  = tid >> 5;
    int lane = threadIdx.x & 31;
    int w    = threadIdx.x >> 5;
    int base = wid * 16;
    bool active = base < N;                       // wave-uniform
    int  col = lane & 15;
    bool hig = (lane & 16) != 0;
    int  kk  = hig ? 8 : 0;

    if (active) {
        int node = base + col; if (node >= N) node = N - 1;
        float dg = deg[node];
        float id_ = invd[node], saV = sa[node], sbV = sb[node];
        float hasf = dg > 0.f ? 1.f : 0.f;
        const float* hrow = hcur + (long)node * 64;
        const float* srow = sums + (long)node * 64;
        const float* qrow = ssq  + (long)node * 64;
        const float* xrow = mxb  + (long)node * 64;
        const float* nrow = mnb  + (long)node * 64;

        v8f acc[4] = {};
        #pragma unroll
        for (int c = 0; c < 26; ++c) {            // K = 832 = 26 chunks of 32
            float f[16];
            build8(f,     32 * c + kk,      hrow, srow, qrow, xrow, nrow, id_, saV, sbV, hasf);
            build8(f + 8, 32 * c + kk + 16, hrow, srow, qrow, xrow, nrow, id_, saV, sbV, hasf);
            v16h a;
            #pragma unroll
            for (int i = 0; i < 16; ++i) a[i] = (_Float16)f[i];
            #pragma unroll
            for (int t = 0; t < 4; ++t) {
                v16h b = *(const v16h*)(pbU + (((c << 2) | t) * 32 + lane) * 16);
                acc[t] = __builtin_amdgcn_wmma_f32_16x16x32_f16(
                    false, a, false, b, (short)0, acc[t], false, false);
            }
        }
        // stage out1 (+bias) to LDS for re-layout (D layout -> A layout)
        #pragma unroll
        for (int t = 0; t < 4; ++t) {
            float ubv = ub[t * 16 + col];
            #pragma unroll
            for (int r = 0; r < 8; ++r) {
                int m = r + (hig ? 8 : 0);
                lds[w * 1024 + m * 64 + t * 16 + col] = (_Float16)(acc[t][r] + ubv);
            }
        }
    }
    __syncthreads();
    if (active) {
        const _Float16* myrow = lds + w * 1024 + col * 64;  // A row M = lane&15
        v8f acc2[4] = {};
        #pragma unroll
        for (int c = 0; c < 2; ++c) {             // K = 64 = 2 chunks of 32
            v8h lo = *(const v8h*)(myrow + 32 * c + kk);
            v8h hi = *(const v8h*)(myrow + 32 * c + kk + 16);
            v16h a = wmma_cat(lo, hi);
            #pragma unroll
            for (int t = 0; t < 4; ++t) {
                v16h b = *(const v16h*)(bXs + (((c << 2) | t) * 32 + lane) * 16);
                acc2[t] = __builtin_amdgcn_wmma_f32_16x16x32_f16(
                    false, a, false, b, (short)0, acc2[t], false, false);
            }
        }
        #pragma unroll
        for (int r = 0; r < 8; ++r) {
            int n2 = base + r + (hig ? 8 : 0);
            if (n2 >= N) continue;
            #pragma unroll
            for (int t = 0; t < 4; ++t) {
                int   cn = t * 16 + col;
                float v  = leaky(acc2[t][r] + xb[cn]) + hcur[(long)n2 * 64 + cn];
                hnext[(long)n2 * 64 + cn] = fmaxf(v, 0.f);    // residual + outer relu
            }
        }
    }
}

// -------------------------------- sum pooling -------------------------------
__global__ void k_pool(const float* __restrict__ hf, float* __restrict__ out, int N) {
    __shared__ float red[256];
    int tid = threadIdx.x;
    int c = tid & 63, rg = tid >> 6;
    float s = 0.f;
    for (long r = (long)blockIdx.x * 4 + rg; r < N; r += (long)gridDim.x * 4)
        s += hf[r * 64 + c];
    red[tid] = s;
    __syncthreads();
    if (tid < 64)
        atomicAdd(out + tid, red[tid] + red[tid + 64] + red[tid + 128] + red[tid + 192]);
}

// --------------------------------- launcher ---------------------------------
extern "C" void kernel_launch(void* const* d_in, const int* in_sizes, int n_in,
                              void* d_out, int out_size, void* d_ws, size_t ws_size,
                              hipStream_t stream) {
    const float* h_in = (const float*)d_in[0];
    const int*   src  = (const int*)d_in[1];
    const int*   dst  = (const int*)d_in[2];
    const float* Mw   = (const float*)d_in[3];
    const float* Mb   = (const float*)d_in[4];
    const float* Uw   = (const float*)d_in[5];
    const float* Ub   = (const float*)d_in[6];
    const float* Xw   = (const float*)d_in[7];
    const float* Xb   = (const float*)d_in[8];
    float* out = (float*)d_out;

    const int N = in_sizes[0] / 64;
    const int E = in_sizes[1];
    const int n64 = N * 64;

    char* ws = (char*)d_ws;
    size_t off = 0;
    auto alloc = [&](size_t bytes) { size_t o = off; off = (off + bytes + 255) & ~(size_t)255; return o; };
    float*    hA   = (float*)(ws + alloc((size_t)n64 * 4));
    float*    hB   = (float*)(ws + alloc((size_t)n64 * 4));
    _Float16* h16  = (_Float16*)(ws + alloc((size_t)n64 * 2));
    float*    sum_ = (float*)(ws + alloc((size_t)n64 * 4));
    float*    ssq_ = (float*)(ws + alloc((size_t)n64 * 4));
    float*    mx_  = (float*)(ws + alloc((size_t)n64 * 4));
    float*    mn_  = (float*)(ws + alloc((size_t)n64 * 4));
    float*    deg_ = (float*)(ws + alloc((size_t)N * 4));
    float*    invd = (float*)(ws + alloc((size_t)N * 4));
    float*    sa_  = (float*)(ws + alloc((size_t)N * 4));
    float*    sb_  = (float*)(ws + alloc((size_t)N * 4));
    _Float16* pbM  = (_Float16*)(ws + alloc((size_t)3 * 4  * 4 * 32 * 16 * 2));
    _Float16* pbU  = (_Float16*)(ws + alloc((size_t)3 * 26 * 4 * 32 * 16 * 2));
    _Float16* pbX  = (_Float16*)(ws + alloc((size_t)3 * 2  * 4 * 32 * 16 * 2));
    (void)ws_size; (void)n_in; (void)out_size;

    const int B256 = 256;
    k_round<<<(n64 + 255) / 256, B256, 0, stream>>>(h_in, hA, n64);
    k_fill<<<(N + 255) / 256, B256, 0, stream>>>(deg_, 0.f, N);
    k_deg<<<(E + 255) / 256, B256, 0, stream>>>(dst, deg_, E);
    k_scal<<<(N + 255) / 256, B256, 0, stream>>>(deg_, invd, sa_, sb_, N);
    k_pack<<<(3 * 32 * 4 * 32 + 255) / 256, B256, 0, stream>>>(Mw, Uw, Xw, pbM, pbU, pbX);

    float* cur = hA;
    float* nxt = hB;
    for (int l = 0; l < 3; ++l) {
        k_layer_init<<<(n64 + 255) / 256, B256, 0, stream>>>(cur, h16, sum_, ssq_, mx_, mn_, n64);
        int ewaves = (E + 15) / 16;
        k_edge<<<(ewaves * 32 + 255) / 256, B256, 0, stream>>>(
            src, dst, h16, pbM + (size_t)l * 4 * 4 * 32 * 16, Mb + l * 64,
            sum_, ssq_, mx_, mn_, E);
        int nwaves = (N + 15) / 16;
        k_node<<<(nwaves * 32 + 255) / 256, B256, 0, stream>>>(
            cur, nxt, sum_, ssq_, mx_, mn_, deg_, invd, sa_, sb_,
            pbU + (size_t)l * 26 * 4 * 32 * 16, pbX + (size_t)l * 2 * 4 * 32 * 16,
            Ub + l * 64, Xb + l * 64, N);
        float* tmp = cur; cur = nxt; nxt = tmp;
    }
    k_fill<<<1, 64, 0, stream>>>(out, 0.f, 64);
    k_pool<<<128, B256, 0, stream>>>(cur, out, N);
}
